// CosineSimAttention_72258529788250
// MI455X (gfx1250) — matrix operand order repaired and checked by reading, not compiled, not run on hardware
//
#include <hip/hip_runtime.h>
#include <hip/hip_bf16.h>
#include <math.h>

// ---------------------------------------------------------------------------
// CosineSimAttention for MI455X (gfx1250, wave32, WMMA + TDM)
//   B=4, S=2048, D=768, H=12, HD=64
// ---------------------------------------------------------------------------

#define BB 4
#define SS 2048
#define DD 768
#define HD 64
#define HH 12
#define EPSN 1e-12f

typedef __attribute__((ext_vector_type(16))) __bf16 v16bf;
typedef __attribute__((ext_vector_type(8)))  __bf16 v8bf;
typedef __attribute__((ext_vector_type(8)))  float  v8f;
typedef __attribute__((ext_vector_type(4)))  unsigned int v4u;
typedef __attribute__((ext_vector_type(8)))  int    v8i;
typedef __attribute__((ext_vector_type(4)))  int    v4i;

#define WMMA_BF16(a, b, c) \
    __builtin_amdgcn_wmma_f32_16x16x32_bf16(false, (a), false, (b), (short)0, (c), false, false)

static __device__ __forceinline__ v16bf mk16(v8bf lo, v8bf hi) {
    v16bf r;
#pragma unroll
    for (int e = 0; e < 8; ++e) { r[e] = lo[e]; r[8 + e] = hi[e]; }
    return r;
}

// LDS byte offset of a __shared__ object (flat LDS aperture keeps offset in [31:0])
static __device__ __forceinline__ unsigned int lds_off(const void* p) {
    return (unsigned int)(uintptr_t)p;
}

// ---------------------------------------------------------------------------
// Tensor Data Mover: 2D bf16 tile Global->LDS with row padding.
//   tile_d0 elements per row (dim0, contiguous), tile_d1 rows,
//   global row stride = stride0 elements, LDS rows padded via pad codes:
//   pad after (2<<pad_int_code) DWORDs, pad amount (pad_amt_code+1) DWORDs.
// Descriptor packed per CDNA5 ISA D# group0/group1 (2D tensor, groups 2/3 zero).
// ---------------------------------------------------------------------------
static __device__ __forceinline__ void tdm_load_2d_bf16(
    unsigned int lds_addr, const void* gptr,
    unsigned int tensor_d0, unsigned int tensor_d1, unsigned long long stride0,
    unsigned int tile_d0, unsigned int tile_d1,
    unsigned int pad_int_code, unsigned int pad_amt_code)
{
    const unsigned long long ga = (unsigned long long)(uintptr_t)gptr;
    v4u g0;
    g0[0] = 1u;                                                     // count=1
    g0[1] = lds_addr;                                               // lds_addr[31:0]
    g0[2] = (unsigned int)ga;                                       // global_addr[31:0]
    g0[3] = (unsigned int)((ga >> 32) & 0x01FFFFFFu) | (2u << 30);  // [56:32] | type=2
    v8i g1;
    g1[0] = (int)((1u << 16)                 // data_size = 1 -> 2 bytes
                | (1u << 20)                 // pad_enable
                | (pad_int_code << 22)       // pad_interval
                | (pad_amt_code << 25));     // pad_amount
    g1[1] = (int)((tensor_d0 & 0xFFFFu) << 16);                       // dim0[15:0]
    g1[2] = (int)((tensor_d0 >> 16) | ((tensor_d1 & 0xFFFFu) << 16)); // dim0[31:16], dim1[15:0]
    g1[3] = (int)((tensor_d1 >> 16) | ((tile_d0 & 0xFFFFu) << 16));   // dim1[31:16], tile_d0
    g1[4] = (int)(tile_d1 & 0xFFFFu);                                 // tile_d1 (tile_d2=0)
    g1[5] = (int)(unsigned int)stride0;                               // dim0_stride[31:0]
    g1[6] = (int)(unsigned int)((stride0 >> 32) & 0xFFFFu);           // dim0_stride[47:32]
    g1[7] = 0;
    v4i z4 = {0, 0, 0, 0};
#if __has_include(<hip/amd_detail/amd_gfx1250_TDM.h>)
    v8i z8 = {0, 0, 0, 0, 0, 0, 0, 0};
    __builtin_amdgcn_tensor_load_to_lds(g0, g1, z4, z4, z8, 0);
#else
    __builtin_amdgcn_tensor_load_to_lds(g0, g1, z4, z4, 0);
#endif
}

// ---------------------------------------------------------------------------
// Kernel 1: fp32 -> bf16 elementwise convert
// ---------------------------------------------------------------------------
__global__ void cvt_bf16_kernel(const float* __restrict__ src, __bf16* __restrict__ dst, int n) {
    int i = blockIdx.x * blockDim.x + threadIdx.x;
    if (i < n) dst[i] = (__bf16)src[i];
}

// ---------------------------------------------------------------------------
// Kernel 2: C[M,N] = A[M,K] @ W[N,K]^T  (torch linear), bf16 WMMA, f32 out.
// Block: 256 threads (8 waves), tile 128(M) x 64(N), K-step 32.
// Tiles staged into LDS by the Tensor Data Mover (wave 0 issues descriptors).
// ---------------------------------------------------------------------------
__global__ __launch_bounds__(256) void gemm_xwT_kernel(const __bf16* __restrict__ A,
                                                       const __bf16* __restrict__ W,
                                                       float* __restrict__ C,
                                                       int M, int N, int K) {
    __shared__ __bf16 As[128][40];   // 32 data + 8 pad (TDM pad: 16DW data, 4DW pad)
    __shared__ __bf16 Bs[64][40];

    const int t    = threadIdx.x;
    const int wave = t >> 5;
    const int lane = t & 31;
    const int hl   = lane >> 4;
    const int l16  = lane & 15;
    const int wm   = wave & 3;
    const int wn   = wave >> 2;
    const int m0   = blockIdx.x * 128;
    const int n0   = blockIdx.y * 64;

    v8f acc[2][2];
#pragma unroll
    for (int i = 0; i < 2; ++i)
#pragma unroll
        for (int j = 0; j < 2; ++j)
#pragma unroll
            for (int r = 0; r < 8; ++r) acc[i][j][r] = 0.0f;

    for (int k0 = 0; k0 < K; k0 += 32) {
        __syncthreads();               // previous tile fully consumed
        if (wave == 0) {
            // A tile: 128 rows x 32 elems, row stride K; pad 16DW/4DW -> As[.][40]
            tdm_load_2d_bf16(lds_off(&As[0][0]), A + (size_t)m0 * K + k0,
                             (unsigned)(K - k0), (unsigned)(M - m0), (unsigned long long)K,
                             32u, 128u, 3u, 3u);
            // W tile: 64 rows x 32 elems
            tdm_load_2d_bf16(lds_off(&Bs[0][0]), W + (size_t)n0 * K + k0,
                             (unsigned)(K - k0), (unsigned)(N - n0), (unsigned long long)K,
                             32u, 64u, 3u, 3u);
            __builtin_amdgcn_s_wait_tensorcnt(0);
        }
        __syncthreads();

        // A fragments (16x32): lanes {L,L+16} -> row L%16, K {0..7,16..23}/{8..15,24..31}
        v16bf a[2];
#pragma unroll
        for (int i = 0; i < 2; ++i) {
            const int row = wm * 32 + i * 16 + l16;
            v8bf lo = *(const v8bf*)&As[row][hl * 8];
            v8bf hi = *(const v8bf*)&As[row][16 + hl * 8];
            a[i] = mk16(lo, hi);
        }
        // B fragments (32x16): lane n -> column n, K {0..7,16..23}/{8..15,24..31}
        v16bf b[2];
#pragma unroll
        for (int j = 0; j < 2; ++j) {
            const int col = wn * 32 + j * 16 + l16;
            v8bf lo = *(const v8bf*)&Bs[col][hl * 8];
            v8bf hi = *(const v8bf*)&Bs[col][16 + hl * 8];
            b[j] = mk16(lo, hi);
        }
#pragma unroll
        for (int i = 0; i < 2; ++i)
#pragma unroll
            for (int j = 0; j < 2; ++j)
                acc[i][j] = WMMA_BF16(a[i], b[j], acc[i][j]);
    }

    // C/D layout: VGPR r, lane half hl -> row r + 8*hl, col = lane%16
#pragma unroll
    for (int i = 0; i < 2; ++i)
#pragma unroll
        for (int j = 0; j < 2; ++j)
#pragma unroll
            for (int r = 0; r < 8; ++r) {
                const int row = m0 + wm * 32 + i * 16 + r + hl * 8;
                const int col = n0 + wn * 32 + j * 16 + l16;
                C[(size_t)row * N + col] = acc[i][j][r];
            }
}

// ---------------------------------------------------------------------------
// Kernel 3: split heads [B*S, D] -> [B,H,S,HD] bf16, optional cosine-normalize
// ---------------------------------------------------------------------------
__global__ void headprep_kernel(const float* __restrict__ src, __bf16* __restrict__ dst,
                                int donorm) {
    const int tid = blockIdx.x * blockDim.x + threadIdx.x;   // (b*H + h)*S + s
    if (tid >= BB * HH * SS) return;
    const int s = tid % SS;
    const int h = (tid / SS) % HH;
    const int b = tid / (SS * HH);
    const float* p = src + (size_t)(b * SS + s) * DD + h * HD;
    float inv = 1.0f;
    if (donorm) {
        float ss = 0.0f;
#pragma unroll 16
        for (int d = 0; d < HD; ++d) { const float v = p[d]; ss += v * v; }
        inv = 1.0f / fmaxf(sqrtf(ss), EPSN);
    }
    __bf16* q = dst + (size_t)tid * HD;
#pragma unroll 16
    for (int d = 0; d < HD; ++d) q[d] = (__bf16)(p[d] * inv);
}

// ---------------------------------------------------------------------------
// Kernel 4: flash attention, causal, per-head temperature.
// Grid: (S/64, H, B).  Block: 128 threads (4 waves); wave w owns 16 query rows.
// K tile staged by TDM; V staged transposed by VALU; next tiles prefetched.
// ---------------------------------------------------------------------------
__global__ __launch_bounds__(128) void attn_kernel(const __bf16* __restrict__ qbf,
                                                   const __bf16* __restrict__ kbf,
                                                   const __bf16* __restrict__ vbf,
                                                   const float* __restrict__ temp,
                                                   __bf16* __restrict__ out) {
    const int qt   = blockIdx.x;          // query tile (64 rows)
    const int h    = blockIdx.y;
    const int b    = blockIdx.z;
    const int t    = threadIdx.x;
    const int wave = t >> 5;
    const int lane = t & 31;
    const int hl   = lane >> 4;
    const int l16  = lane & 15;
    const int qb   = qt * 64;
    const size_t base = (size_t)(b * HH + h) * SS * HD;

    __shared__ __bf16 Ks[64][72];          // keys row-major [key][dim] (TDM pad 32DW+4DW)
    __shared__ __bf16 VTs[64][72];         // V transposed [dim][key]
    __shared__ __bf16 Ps[4][16][72];       // per-wave probability staging (C->A relayout)

    // Q A-fragments (held in registers for the whole kernel)
    v16bf qa[2];
    {
        const int qrow = qb + wave * 16 + l16;
        const __bf16* qp = qbf + base + (size_t)qrow * HD;
#pragma unroll
        for (int kc = 0; kc < 2; ++kc) {
            v8bf lo = *(const v8bf*)(qp + kc * 32 + hl * 8);
            v8bf hi = *(const v8bf*)(qp + kc * 32 + 16 + hl * 8);
            qa[kc] = mk16(lo, hi);
        }
    }
    const float tscale = temp[h];

    v8f   o[4];
    float mrow[8], lrow[8];
#pragma unroll
    for (int dt = 0; dt < 4; ++dt)
#pragma unroll
        for (int r = 0; r < 8; ++r) o[dt][r] = 0.0f;
#pragma unroll
    for (int r = 0; r < 8; ++r) { mrow[r] = -INFINITY; lrow[r] = 0.0f; }

    for (int jt = 0; jt <= qt; ++jt) {
        const int jb = jt * 64;
        __syncthreads();                   // previous iteration readers done
        // K tile via Tensor Data Mover: 64x64, row stride HD, pad to 72-elem rows
        if (wave == 0)
            tdm_load_2d_bf16(lds_off(&Ks[0][0]), kbf + base + (size_t)jb * HD,
                             (unsigned)HD, (unsigned)(SS - jb), (unsigned long long)HD,
                             64u, 64u, 4u, 3u);
        // V tile transposed into LDS (each thread 32 elems)
        {
            const int row = t >> 1, kk = (t & 1) * 32;
            const __bf16* gv = vbf + base + (size_t)(jb + row) * HD + kk;
#pragma unroll
            for (int c = 0; c < 32; ++c) VTs[kk + c][row] = gv[c];
        }
        // prefetch next iteration's K/V tiles into cache
        if (jt < qt) {
            const __bf16* nk = kbf + base + (size_t)(jb + 64 + (t >> 1)) * HD;
            const __bf16* nv = vbf + base + (size_t)(jb + 64 + (t >> 1)) * HD;
            __builtin_prefetch(nk, 0, 3);
            __builtin_prefetch(nv, 0, 3);
        }
        if (wave == 0) __builtin_amdgcn_s_wait_tensorcnt(0);
        __syncthreads();

        // S = Q @ K^T : 4 n-tiles of 16 keys, 2 K-chunks of 32 dims
        v8f s[4];
#pragma unroll
        for (int nt = 0; nt < 4; ++nt)
#pragma unroll
            for (int r = 0; r < 8; ++r) s[nt][r] = 0.0f;
#pragma unroll
        for (int kc = 0; kc < 2; ++kc)
#pragma unroll
            for (int nt = 0; nt < 4; ++nt) {
                const int col = nt * 16 + l16;
                v8bf lo = *(const v8bf*)&Ks[col][kc * 32 + hl * 8];
                v8bf hi = *(const v8bf*)&Ks[col][kc * 32 + 16 + hl * 8];
                s[nt] = WMMA_BF16(qa[kc], mk16(lo, hi), s[nt]);
            }

        // temperature + causal mask + row max (lane-local, then 16-lane butterfly)
        const bool diag = (jt == qt);
        float lm[8];
#pragma unroll
        for (int r = 0; r < 8; ++r) lm[r] = -INFINITY;
#pragma unroll
        for (int nt = 0; nt < 4; ++nt) {
            const int key = jb + nt * 16 + l16;
#pragma unroll
            for (int r = 0; r < 8; ++r) {
                float v = s[nt][r] * tscale;
                const int qi = qb + wave * 16 + r + hl * 8;
                if (diag && key > qi) v = -INFINITY;
                s[nt][r] = v;
                lm[r] = fmaxf(lm[r], v);
            }
        }
#pragma unroll
        for (int off = 1; off < 16; off <<= 1)
#pragma unroll
            for (int r = 0; r < 8; ++r) lm[r] = fmaxf(lm[r], __shfl_xor(lm[r], off, 32));

        // online softmax rescale
        float ls[8];
#pragma unroll
        for (int r = 0; r < 8; ++r) {
            const float mnew = fmaxf(mrow[r], lm[r]);
            const float sc = __expf(mrow[r] - mnew);
            mrow[r] = mnew;
            lrow[r] *= sc;
#pragma unroll
            for (int dt = 0; dt < 4; ++dt) o[dt][r] *= sc;
            ls[r] = 0.0f;
        }
        // exp, partial row-sums, C->A relayout of P via per-wave LDS
#pragma unroll
        for (int nt = 0; nt < 4; ++nt)
#pragma unroll
            for (int r = 0; r < 8; ++r) {
                const float p = __expf(s[nt][r] - mrow[r]);
                ls[r] += p;
                Ps[wave][r + hl * 8][nt * 16 + l16] = (__bf16)p;
            }
#pragma unroll
        for (int off = 1; off < 16; off <<= 1)
#pragma unroll
            for (int r = 0; r < 8; ++r) ls[r] += __shfl_xor(ls[r], off, 32);
#pragma unroll
        for (int r = 0; r < 8; ++r) lrow[r] += ls[r];

        // O += P @ V   (P: 16x64 A-frags from LDS; V^T: contiguous B-frags)
#pragma unroll
        for (int kc = 0; kc < 2; ++kc) {
            const __bf16* pp = &Ps[wave][l16][0];
            v8bf plo = *(const v8bf*)(pp + kc * 32 + hl * 8);
            v8bf phi = *(const v8bf*)(pp + kc * 32 + 16 + hl * 8);
            const v16bf pa = mk16(plo, phi);
#pragma unroll
            for (int dt = 0; dt < 4; ++dt) {
                const int dcol = dt * 16 + l16;
                v8bf lo = *(const v8bf*)&VTs[dcol][kc * 32 + hl * 8];
                v8bf hi = *(const v8bf*)&VTs[dcol][kc * 32 + 16 + hl * 8];
                o[dt] = WMMA_BF16(pa, mk16(lo, hi), o[dt]);
            }
        }
    }

    // epilogue: divide by row sums, store bf16 [B,S,H,HD]
#pragma unroll
    for (int r = 0; r < 8; ++r) {
        const float inv = (lrow[r] > 0.0f) ? 1.0f / lrow[r] : 0.0f;
        const int q = qb + wave * 16 + r + hl * 8;
        __bf16* op = out + ((size_t)(b * SS + q) * HH + h) * HD;
#pragma unroll
        for (int dt = 0; dt < 4; ++dt)
            op[dt * 16 + l16] = (__bf16)(o[dt][r] * inv);
    }
}

// ---------------------------------------------------------------------------
// Host-side launcher
// ---------------------------------------------------------------------------
extern "C" void kernel_launch(void* const* d_in, const int* in_sizes, int n_in,
                              void* d_out, int out_size, void* d_ws, size_t ws_size,
                              hipStream_t stream) {
    (void)in_sizes; (void)n_in; (void)out_size; (void)ws_size;
    const float* x    = (const float*)d_in[0];
    const float* Wq   = (const float*)d_in[1];
    const float* Wk   = (const float*)d_in[2];
    const float* Wv   = (const float*)d_in[3];
    const float* Wo   = (const float*)d_in[4];
    const float* temp = (const float*)d_in[5];

    const int xel = BB * SS * DD;          // 6,291,456
    const int wel = DD * DD;               // 589,824
    char* ws = (char*)d_ws;
    size_t off = 0;
    __bf16* xbf = (__bf16*)(ws + off); off += (size_t)xel * 2;                   // 12.58 MB
    __bf16* wbf[4];
    for (int i = 0; i < 4; ++i) { wbf[i] = (__bf16*)(ws + off); off += (size_t)wel * 2; }
    float*  tmpf = (float*)(ws + off); off += (size_t)xel * 4;                   // reused Q->K->V
    __bf16* qbf  = (__bf16*)(ws + off); off += (size_t)xel * 2;
    __bf16* kbf  = (__bf16*)(ws + off); off += (size_t)xel * 2;
    __bf16* vbf  = (__bf16*)(ws + off); off += (size_t)xel * 2;
    __bf16* abf  = (__bf16*)(ws + off); off += (size_t)xel * 2;                  // total ~92.8 MB

    // 1) convert inputs to bf16
    cvt_bf16_kernel<<<(xel + 255) / 256, 256, 0, stream>>>(x, xbf, xel);
    cvt_bf16_kernel<<<(wel + 255) / 256, 256, 0, stream>>>(Wq, wbf[0], wel);
    cvt_bf16_kernel<<<(wel + 255) / 256, 256, 0, stream>>>(Wk, wbf[1], wel);
    cvt_bf16_kernel<<<(wel + 255) / 256, 256, 0, stream>>>(Wv, wbf[2], wel);
    cvt_bf16_kernel<<<(wel + 255) / 256, 256, 0, stream>>>(Wo, wbf[3], wel);

    const dim3 gGemm(BB * SS / 128, DD / 64);      // 64 x 12
    const int  nvec = BB * HH * SS;                // 98,304 head-vectors
    const dim3 gPrep((nvec + 255) / 256);

    // 2) projections + head split/normalize (fp32 scratch reused)
    gemm_xwT_kernel<<<gGemm, 256, 0, stream>>>(xbf, wbf[0], tmpf, BB * SS, DD, DD);
    headprep_kernel<<<gPrep, 256, 0, stream>>>(tmpf, qbf, 1);
    gemm_xwT_kernel<<<gGemm, 256, 0, stream>>>(xbf, wbf[1], tmpf, BB * SS, DD, DD);
    headprep_kernel<<<gPrep, 256, 0, stream>>>(tmpf, kbf, 1);
    gemm_xwT_kernel<<<gGemm, 256, 0, stream>>>(xbf, wbf[2], tmpf, BB * SS, DD, DD);
    headprep_kernel<<<gPrep, 256, 0, stream>>>(tmpf, vbf, 0);

    // 3) causal flash attention
    attn_kernel<<<dim3(SS / 64, HH, BB), 128, 0, stream>>>(qbf, kbf, vbf, temp, abf);

    // 4) output projection straight into d_out (fp32)
    gemm_xwT_kernel<<<gGemm, 256, 0, stream>>>(abf, wbf[3], (float*)d_out, BB * SS, DD, DD);
}
